// SwinTransformerBlock_60481729463048
// MI455X (gfx1250) — matrix operand order repaired and checked by reading, not compiled, not run on hardware
//
#include <hip/hip_runtime.h>
#include <hip/hip_bf16.h>
#include <math.h>
#include <stdint.h>

typedef __bf16 bf16_t;
typedef __attribute__((ext_vector_type(16))) __bf16 v16bf;
typedef __attribute__((ext_vector_type(8)))  __bf16 v8bf;
typedef __attribute__((ext_vector_type(8)))  float  v8f;
typedef __attribute__((ext_vector_type(4)))  unsigned int v4u;
typedef __attribute__((ext_vector_type(4)))  int v4i;
typedef __attribute__((ext_vector_type(8)))  int v8i;

#define WMMA_BF16(A, B, C) \
  __builtin_amdgcn_wmma_f32_16x16x32_bf16(false, (A), false, (B), (short)0, (C), false, false)

// ---- CDNA5 feature probes (device pass only; host pass gets fallbacks) ----
#if defined(__HIP_DEVICE_COMPILE__) && __has_builtin(__builtin_amdgcn_tensor_load_to_lds)
#define USE_TDM 1
#else
#define USE_TDM 0
#endif

#if defined(__HIP_DEVICE_COMPILE__) && __has_builtin(__builtin_amdgcn_global_load_async_to_lds_b128)
#define USE_ASYNC 1
#else
#define USE_ASYNC 0
#endif

#if defined(__HIP_DEVICE_COMPILE__)
#if USE_TDM
#warning "CDNA5 probe: tensor_load_to_lds builtin AVAILABLE (TDM path enabled)"
#else
#warning "CDNA5 probe: tensor_load_to_lds builtin NOT available (fallback staging)"
#endif
#if USE_ASYNC
#warning "CDNA5 probe: global_load_async_to_lds_b128 builtin AVAILABLE (async path enabled)"
#else
#warning "CDNA5 probe: global_load_async_to_lds_b128 builtin NOT available"
#endif
#endif

#if USE_TDM
__device__ inline void wait_tensorcnt0() {
#if __has_builtin(__builtin_amdgcn_s_wait_tensorcnt)
  __builtin_amdgcn_s_wait_tensorcnt(0);
#else
  asm volatile("s_wait_tensorcnt 0x0" ::: "memory");
#endif
}

// Issue a TDM 2D tile load: tile_h rows of tile_w bf16 elements, global row
// stride row_stride_elems, written CONTIGUOUSLY to LDS (matches unpadded tile).
// D# packing per CDNA5 ISA ch.8 (group0: count/lds/global/type, group1: dims).
__device__ inline void tdm_load_tile(const bf16_t* gsrc, const bf16_t* ldst,
                                     int tile_w, int tile_h, int row_stride_elems) {
  unsigned long long ga = (unsigned long long)(uintptr_t)gsrc;
  unsigned lds = (unsigned)(uintptr_t)ldst;  // low 32 bits of LDS-aperture addr
  v4u g0;
  g0[0] = 1u;                                   // count=1, user descriptor
  g0[1] = lds;                                  // lds_addr [63:32]
  g0[2] = (unsigned)ga;                         // global_addr lo [95:64]
  g0[3] = (unsigned)((ga >> 32) & 0x01ffffffULL) | 0x80000000u;  // ga hi | type=2
  v8i g1;
  g1[0] = (int)(1u << 16);                      // data_size=1 (2 bytes)
  g1[1] = (int)(((unsigned)tile_w) << 16);      // tensor_dim0[15:0] @ [63:48]
  g1[2] = (int)(((unsigned)tile_h) << 16);      // tensor_dim0 hi=0 | tensor_dim1 lo
  g1[3] = (int)(((unsigned)tile_w) << 16);      // tensor_dim1 hi=0 | tile_dim0
  g1[4] = (int)((unsigned)tile_h);              // tile_dim1 | tile_dim2=0
  g1[5] = row_stride_elems;                     // tensor_dim0_stride lo32
  g1[6] = 0;
  g1[7] = 0;
  v4i zz = {0, 0, 0, 0};
#if __clang_major__ >= 23
  v8i z8 = {0, 0, 0, 0, 0, 0, 0, 0};
  __builtin_amdgcn_tensor_load_to_lds(g0, g1, zz, zz, z8, 0);
#else
  __builtin_amdgcn_tensor_load_to_lds(g0, g1, zz, zz, 0);
#endif
}
#endif  // USE_TDM

#if USE_ASYNC
// 16B global -> LDS async copy (ASYNCcnt). Builtin wants v4i* in AS(1)/AS(3);
// build pointers via inttoptr (always legal).
typedef __attribute__((address_space(1))) v4i as1_v4i;
typedef __attribute__((address_space(3))) v4i as3_v4i;
__device__ inline void async_cp16(const void* g, void* l) {
  __builtin_amdgcn_global_load_async_to_lds_b128(
      (as1_v4i*)(uintptr_t)g,
      (as3_v4i*)(unsigned)(uintptr_t)l,
      0, 0);
}
__device__ inline void wait_asynccnt0() {
#if __has_builtin(__builtin_amdgcn_s_wait_asynccnt)
  __builtin_amdgcn_s_wait_asynccnt(0);
#else
  asm volatile("s_wait_asynccnt 0x0" ::: "memory");
#endif
}
#endif  // USE_ASYNC

// ---------------------------------------------------------------------------
// f32 -> bf16 weight conversion
// ---------------------------------------------------------------------------
__global__ void __launch_bounds__(256) f32_to_bf16_kernel(const float* __restrict__ in,
                                                          bf16_t* __restrict__ out, long n) {
  long i = (long)blockIdx.x * blockDim.x + threadIdx.x;
  if (i < n) out[i] = (bf16_t)in[i];
}

// ---------------------------------------------------------------------------
// LayerNorm (512 wide), optional window-partition permutation, f32 -> bf16
// ---------------------------------------------------------------------------
template <int WINDOWED>
__global__ void __launch_bounds__(128) ln_kernel(const float* __restrict__ x,
                                                 const float* __restrict__ g,
                                                 const float* __restrict__ bta,
                                                 bf16_t* __restrict__ out) {
  long orow = blockIdx.x;  // 0..65535
  long irow;
  if (WINDOWED) {
    long b_ = orow >> 6;
    int  wn = (int)(orow & 63);
    long b  = b_ >> 6;
    int  wy = (int)((b_ >> 3) & 7), wx = (int)(b_ & 7);
    int  y  = wy * 8 + (wn >> 3);
    int  xc = wx * 8 + (wn & 7);
    irow = b * 4096 + (long)y * 64 + xc;
  } else {
    irow = orow;
  }
  int tid = threadIdx.x;
  const float4* xr = (const float4*)(x + irow * 512);
  float4 v = xr[tid];
  float s  = v.x + v.y + v.z + v.w;
  float s2 = v.x * v.x + v.y * v.y + v.z * v.z + v.w * v.w;
#pragma unroll
  for (int o = 1; o < 32; o <<= 1) {
    s  += __shfl_xor(s, o, 32);
    s2 += __shfl_xor(s2, o, 32);
  }
  __shared__ float sh[8];
  int wv = tid >> 5, ln = tid & 31;
  if (ln == 0) { sh[wv] = s; sh[4 + wv] = s2; }
  __syncthreads();
  float ts   = sh[0] + sh[1] + sh[2] + sh[3];
  float ts2  = sh[4] + sh[5] + sh[6] + sh[7];
  float mean = ts * (1.0f / 512.0f);
  float var  = ts2 * (1.0f / 512.0f) - mean * mean;
  float rstd = rsqrtf(var + 1e-5f);
  float4 gg = ((const float4*)g)[tid];
  float4 bb = ((const float4*)bta)[tid];
  bf16_t* op = out + orow * 512 + tid * 4;
  op[0] = (bf16_t)((v.x - mean) * rstd * gg.x + bb.x);
  op[1] = (bf16_t)((v.y - mean) * rstd * gg.y + bb.y);
  op[2] = (bf16_t)((v.z - mean) * rstd * gg.z + bb.z);
  op[3] = (bf16_t)((v.w - mean) * rstd * gg.w + bb.w);
}

// ---------------------------------------------------------------------------
// WMMA GEMM: C[M,N] = A[M,K](bf16,rm) * W[N,K](bf16,rm)^T
// Block = 256 thr (8 waves); tile 128x64, BK=64; each wave owns 16 rows x 64
// cols => 8 WMMA / wave / K-step. Staging priority: TDM > async-LDS > vector.
// EPI: 0=QKV scatter, 1=PROJ(+x residual, window reverse -> f32),
//      2=FC1(+GELU -> bf16), 3=FC2(+residual -> f32)
// ---------------------------------------------------------------------------
template <int EPI>
__global__ void __launch_bounds__(256) gemm_wmma(
    const bf16_t* __restrict__ A, const bf16_t* __restrict__ W,
    const float* __restrict__ bias, int K,
    float* __restrict__ outF, bf16_t* __restrict__ outB,
    bf16_t* __restrict__ outB2, bf16_t* __restrict__ outB3,
    const float* __restrict__ res) {
  __shared__ __align__(16) bf16_t Als[2][128][64];
  __shared__ __align__(16) bf16_t Bls[2][64][64];

  const int tid  = threadIdx.x;
  const int wave = tid >> 5;
  const int lane = tid & 31;
  const int hf   = lane >> 4;  // lane half
  const int r    = lane & 15;
  const long m0  = (long)blockIdx.y * 128;
  const long n0  = (long)blockIdx.x * 64;

  v8f acc[4] = {};
  const int nIter = K >> 6;

#if USE_TDM
  if (wave == 0) {
    tdm_load_tile(A + m0 * K, &Als[0][0][0], 64, 128, K);
    tdm_load_tile(W + n0 * K, &Bls[0][0][0], 64, 64, K);
  }
  for (int i = 0; i < nIter; ++i) {
    const int cur = i & 1;
    if (wave == 0) wait_tensorcnt0();   // current buffer DMA complete (wave0)
    __syncthreads();                    // publish tile; all reads of next buf done
    if (wave == 0 && i + 1 < nIter) {   // overlap next tile DMA with compute
      long k0 = (long)(i + 1) << 6;
      tdm_load_tile(A + m0 * K + k0, &Als[cur ^ 1][0][0], 64, 128, K);
      tdm_load_tile(W + n0 * K + k0, &Bls[cur ^ 1][0][0], 64, 64, K);
    }
#pragma unroll
    for (int ks = 0; ks < 2; ks++) {
      v16bf af;
      const bf16_t* arow = &Als[cur][wave * 16 + r][ks * 32 + hf * 8];
      ((v8bf*)&af)[0] = *(const v8bf*)arow;
      ((v8bf*)&af)[1] = *(const v8bf*)(arow + 16);
#pragma unroll
      for (int nt = 0; nt < 4; nt++) {
        const bf16_t* brow = &Bls[cur][nt * 16 + r][ks * 32 + hf * 8];
        v16bf bv;
        ((v8bf*)&bv)[0] = *(const v8bf*)brow;
        ((v8bf*)&bv)[1] = *(const v8bf*)(brow + 16);
        acc[nt] = WMMA_BF16(af, bv, acc[nt]);
      }
    }
  }
#else
  for (int i = 0; i < nIter; ++i) {
    const long k0 = (long)i << 6;
    __syncthreads();  // previous tile fully consumed before overwrite
#if USE_ASYNC
#pragma unroll
    for (int j = 0; j < 4; j++) {  // A: 128x64 = 1024 16B chunks
      int c = tid + j * 256;
      int row = c >> 3, col = (c & 7) * 8;
      async_cp16(A + (m0 + row) * K + k0 + col, &Als[0][row][col]);
    }
#pragma unroll
    for (int j = 0; j < 2; j++) {  // B: 64x64 = 512 16B chunks
      int c = tid + j * 256;
      int row = c >> 3, col = (c & 7) * 8;
      async_cp16(W + (n0 + row) * K + k0 + col, &Bls[0][row][col]);
    }
    wait_asynccnt0();
#else
#pragma unroll
    for (int j = 0; j < 4; j++) {
      int c = tid + j * 256;
      int row = c >> 3, col = (c & 7) * 8;
      *(v8bf*)&Als[0][row][col] = *(const v8bf*)(A + (m0 + row) * K + k0 + col);
    }
#pragma unroll
    for (int j = 0; j < 2; j++) {
      int c = tid + j * 256;
      int row = c >> 3, col = (c & 7) * 8;
      *(v8bf*)&Bls[0][row][col] = *(const v8bf*)(W + (n0 + row) * K + k0 + col);
    }
    if (i + 1 < nIter) __builtin_prefetch(A + (m0 + (tid >> 3)) * K + k0 + 64, 0, 0);
#endif
    __syncthreads();
#pragma unroll
    for (int ks = 0; ks < 2; ks++) {
      v16bf af;
      const bf16_t* arow = &Als[0][wave * 16 + r][ks * 32 + hf * 8];
      ((v8bf*)&af)[0] = *(const v8bf*)arow;
      ((v8bf*)&af)[1] = *(const v8bf*)(arow + 16);
#pragma unroll
      for (int nt = 0; nt < 4; nt++) {
        const bf16_t* brow = &Bls[0][nt * 16 + r][ks * 32 + hf * 8];
        v16bf bv;
        ((v8bf*)&bv)[0] = *(const v8bf*)brow;
        ((v8bf*)&bv)[1] = *(const v8bf*)(brow + 16);
        acc[nt] = WMMA_BF16(af, bv, acc[nt]);
      }
    }
  }
#endif

#pragma unroll
  for (int nt = 0; nt < 4; nt++) {
#pragma unroll
    for (int e = 0; e < 8; e++) {
      long m = m0 + wave * 16 + e + hf * 8;
      long n = n0 + nt * 16 + r;
      float val = acc[nt][e] + bias[n];
      if (EPI == 0) {  // QKV scatter to per-head layouts; scale q
        int  tq = (int)(n >> 9);
        int  h  = (int)((n >> 5) & 15);
        int  d  = (int)(n & 31);
        long b_ = m >> 6;
        int  wn = (int)(m & 63);
        long qk_idx = ((b_ * 16 + h) * 64 + wn) * 32 + d;   // (B_,NH,N,HD)
        long v_idx  = ((b_ * 16 + h) * 32 + d) * 64 + wn;   // (B_,NH,HD,N)
        if (tq == 0)      outB [qk_idx] = (bf16_t)(val * 0.17677669529663687f);
        else if (tq == 1) outB2[qk_idx] = (bf16_t)val;
        else              outB3[v_idx]  = (bf16_t)val;
      } else if (EPI == 1) {  // PROJ: window reverse + residual -> f32
        long b_ = m >> 6;
        int  wn = (int)(m & 63);
        long b  = b_ >> 6;
        int  wy = (int)((b_ >> 3) & 7), wx = (int)(b_ & 7);
        int  y  = wy * 8 + (wn >> 3);
        int  xc = wx * 8 + (wn & 7);
        long gidx = ((b * 4096 + (long)y * 64 + xc) * 512 + n);
        outF[gidx] = res[gidx] + val;
      } else if (EPI == 2) {  // FC1 + exact GELU -> bf16
        float ge = 0.5f * val * (1.0f + erff(val * 0.7071067811865475f));
        outB[m * 2048 + n] = (bf16_t)ge;
      } else {  // FC2 + residual -> f32 (final output)
        long gidx = m * 512 + n;
        outF[gidx] = res[gidx] + val;
      }
    }
  }
}

// ---------------------------------------------------------------------------
// Fused window attention: one block = (window b_, head h). 128 threads/4 waves.
// ---------------------------------------------------------------------------
__global__ void __launch_bounds__(128) attn_wmma(
    const bf16_t* __restrict__ q, const bf16_t* __restrict__ k,
    const bf16_t* __restrict__ vt, const float* __restrict__ D,
    const float* __restrict__ a_p, const float* __restrict__ b_p,
    const float* __restrict__ a_r, const float* __restrict__ b_r,
    bf16_t* __restrict__ AO) {
  const int h  = blockIdx.x;  // 0..15
  const int b_ = blockIdx.y;  // 0..1023

  __shared__ __align__(16) bf16_t Qs[64][32];
  __shared__ __align__(16) bf16_t Ks[64][32];
  __shared__ __align__(16) bf16_t Vts[32][64];
  __shared__ __align__(16) bf16_t Ps[64][72];
  __shared__ float Dls[64];
  __shared__ float APs[15], BPs[15], ARs[15], BRs[15];

  const int tid = threadIdx.x;
  long base = ((long)b_ * 16 + h) * 2048;
  const v8bf* qg = (const v8bf*)(q + base);
  const v8bf* kg = (const v8bf*)(k + base);
  const v8bf* vg = (const v8bf*)(vt + base);
  v8bf* qs = (v8bf*)&Qs[0][0];
  v8bf* ks = (v8bf*)&Ks[0][0];
  v8bf* vs = (v8bf*)&Vts[0][0];
#if USE_ASYNC
#pragma unroll
  for (int c = tid; c < 256; c += 128) {
    async_cp16(qg + c, qs + c);
    async_cp16(kg + c, ks + c);
    async_cp16(vg + c, vs + c);
  }
#else
#pragma unroll
  for (int c = tid; c < 256; c += 128) { qs[c] = qg[c]; ks[c] = kg[c]; vs[c] = vg[c]; }
#endif

  long b  = b_ >> 6;
  int  wy = (b_ >> 3) & 7, wx = b_ & 7;
  if (tid < 64) {
    int jy = tid >> 3, jx = tid & 7;
    Dls[tid] = D[(b * 64 + wy * 8 + jy) * 64 + wx * 8 + jx];
  }
  if (tid < 15) {
    APs[tid] = a_p[tid * 16 + h];
    BPs[tid] = b_p[tid * 16 + h];
    ARs[tid] = a_r[tid * 16 + h];
    BRs[tid] = b_r[tid * 16 + h];
  }
#if USE_ASYNC
  wait_asynccnt0();
#endif
  __syncthreads();

  const int wave = tid >> 5, lane = tid & 31;
  const int hf = lane >> 4, r = lane & 15;
  const int mt = wave;

  // S = q * k^T
  v16bf aq;
  ((v8bf*)&aq)[0] = *(const v8bf*)&Qs[mt * 16 + r][hf * 8];
  ((v8bf*)&aq)[1] = *(const v8bf*)&Qs[mt * 16 + r][hf * 8 + 16];
  v8f S[4];
#pragma unroll
  for (int nt = 0; nt < 4; nt++) {
    v16bf bk;
    ((v8bf*)&bk)[0] = *(const v8bf*)&Ks[nt * 16 + r][hf * 8];
    ((v8bf*)&bk)[1] = *(const v8bf*)&Ks[nt * 16 + r][hf * 8 + 16];
    v8f z = {};
    S[nt] = WMMA_BF16(aq, bk, z);
  }

  // Bias + row softmax
#pragma unroll
  for (int e = 0; e < 8; e++) {
    int i = mt * 16 + e + hf * 8;
    float sv[4];
#pragma unroll
    for (int nt = 0; nt < 4; nt++) {
      int j  = nt * 16 + r;
      int az = (i & 7) - (j & 7);
      int ip = az % 15; if (ip < 0) ip += 15;
      float azf = (float)az * 0.09817477042468103f;  // 2*pi/64
      int rad = (i >> 3) - (j >> 3);
      int ir  = rad % 15; if (ir < 0) ir += 15;
      float ang = (float)rad * Dls[j] * 0.02454369260617026f;  // 2*pi/256
      sv[nt] = S[nt][e] + APs[ip] * cosf(azf) + BPs[ip] * sinf(azf)
                        + ARs[ir] * cosf(ang) + BRs[ir] * sinf(ang);
    }
    float mx = fmaxf(fmaxf(sv[0], sv[1]), fmaxf(sv[2], sv[3]));
#pragma unroll
    for (int o = 1; o < 16; o <<= 1) mx = fmaxf(mx, __shfl_xor(mx, o, 32));
    float sum = 0.0f;
#pragma unroll
    for (int nt = 0; nt < 4; nt++) { sv[nt] = expf(sv[nt] - mx); sum += sv[nt]; }
#pragma unroll
    for (int o = 1; o < 16; o <<= 1) sum += __shfl_xor(sum, o, 32);
    float inv = 1.0f / sum;
#pragma unroll
    for (int nt = 0; nt < 4; nt++) Ps[i][nt * 16 + r] = (bf16_t)(sv[nt] * inv);
  }
  __syncthreads();

  // O = P * V
  v8f O0 = {}, O1 = {};
#pragma unroll
  for (int ks2 = 0; ks2 < 64; ks2 += 32) {
    v16bf ap;
    ((v8bf*)&ap)[0] = *(const v8bf*)&Ps[mt * 16 + r][ks2 + hf * 8];
    ((v8bf*)&ap)[1] = *(const v8bf*)&Ps[mt * 16 + r][ks2 + hf * 8 + 16];
    v16bf bv0, bv1;
    ((v8bf*)&bv0)[0] = *(const v8bf*)&Vts[r][ks2 + hf * 8];
    ((v8bf*)&bv0)[1] = *(const v8bf*)&Vts[r][ks2 + hf * 8 + 16];
    ((v8bf*)&bv1)[0] = *(const v8bf*)&Vts[16 + r][ks2 + hf * 8];
    ((v8bf*)&bv1)[1] = *(const v8bf*)&Vts[16 + r][ks2 + hf * 8 + 16];
    O0 = WMMA_BF16(ap, bv0, O0);
    O1 = WMMA_BF16(ap, bv1, O1);
  }

  long ob = (long)b_ * 64 * 512 + h * 32;
#pragma unroll
  for (int e = 0; e < 8; e++) {
    int m = mt * 16 + e + hf * 8;
    AO[ob + (long)m * 512 + r]      = (bf16_t)O0[e];
    AO[ob + (long)m * 512 + 16 + r] = (bf16_t)O1[e];
  }
}

// ---------------------------------------------------------------------------
// Host launcher
// ---------------------------------------------------------------------------
extern "C" void kernel_launch(void* const* d_in, const int* in_sizes, int n_in,
                              void* d_out, int out_size, void* d_ws, size_t ws_size,
                              hipStream_t stream) {
  (void)in_sizes; (void)n_in; (void)out_size; (void)ws_size;
  const float* x      = (const float*)d_in[0];
  const float* D      = (const float*)d_in[1];
  const float* n1g    = (const float*)d_in[2];
  const float* n1b    = (const float*)d_in[3];
  const float* qkv_w  = (const float*)d_in[4];
  const float* qkv_b  = (const float*)d_in[5];
  const float* proj_w = (const float*)d_in[6];
  const float* proj_b = (const float*)d_in[7];
  const float* a_p    = (const float*)d_in[8];
  const float* b_p    = (const float*)d_in[9];
  const float* a_r    = (const float*)d_in[10];
  const float* b_r    = (const float*)d_in[11];
  const float* n2g    = (const float*)d_in[12];
  const float* n2b    = (const float*)d_in[13];
  const float* fc1_w  = (const float*)d_in[14];
  const float* fc1_b  = (const float*)d_in[15];
  const float* fc2_w  = (const float*)d_in[16];
  const float* fc2_b  = (const float*)d_in[17];

  char* ws = (char*)d_ws;
  bf16_t* XW  = (bf16_t*)(ws + 0);
  bf16_t* Qb  = (bf16_t*)(ws + 67108864LL);
  bf16_t* Kb  = (bf16_t*)(ws + 134217728LL);
  bf16_t* Vt  = (bf16_t*)(ws + 201326592LL);
  float*  XR  = (float*) (ws + 268435456LL);
  bf16_t* LN2 = (bf16_t*)(ws + 402653184LL);
  bf16_t* WQ  = (bf16_t*)(ws + 469762048LL);
  bf16_t* WP  = WQ + 786432;
  bf16_t* W1  = WP + 262144;
  bf16_t* W2  = W1 + 1048576;
  bf16_t* AO  = (bf16_t*)(ws + 0);   // reuse XW region (free after QKV GEMM)
  bf16_t* Hb  = (bf16_t*)(ws + 0);   // reuse [0,256Mi) (free after attn+proj)

  f32_to_bf16_kernel<<<3072, 256, 0, stream>>>(qkv_w, WQ, 786432);
  f32_to_bf16_kernel<<<1024, 256, 0, stream>>>(proj_w, WP, 262144);
  f32_to_bf16_kernel<<<4096, 256, 0, stream>>>(fc1_w, W1, 1048576);
  f32_to_bf16_kernel<<<4096, 256, 0, stream>>>(fc2_w, W2, 1048576);

  ln_kernel<1><<<65536, 128, 0, stream>>>(x, n1g, n1b, XW);

  gemm_wmma<0><<<dim3(24, 512), 256, 0, stream>>>(XW, WQ, qkv_b, 512,
                                                  nullptr, Qb, Kb, Vt, nullptr);

  attn_wmma<<<dim3(16, 1024), 128, 0, stream>>>(Qb, Kb, Vt, D, a_p, b_p, a_r, b_r, AO);

  gemm_wmma<1><<<dim3(8, 512), 256, 0, stream>>>(AO, WP, proj_b, 512,
                                                 XR, nullptr, nullptr, nullptr, x);

  ln_kernel<0><<<65536, 128, 0, stream>>>(XR, n2g, n2b, LN2);

  gemm_wmma<2><<<dim3(32, 512), 256, 0, stream>>>(LN2, W1, fc1_b, 512,
                                                  nullptr, Hb, nullptr, nullptr, nullptr);

  gemm_wmma<3><<<dim3(8, 512), 256, 0, stream>>>(Hb, W2, fc2_b, 2048,
                                                 (float*)d_out, nullptr, nullptr, nullptr, XR);
}
